// GATMultiHeadBlockUnidirect_13134009991787
// MI455X (gfx1250) — matrix operand order repaired and checked by reading, not compiled, not run on hardware
//
#include <hip/hip_runtime.h>
#include <hip/hip_bf16.h>

// Problem sizes (fixed by the reference)
constexpr int N_NODES = 50000;
constexpr int N_EDGES = 800000;
constexpr int F_DIM   = 64;
constexpr int H_HEADS = 4;
constexpr int HC      = 256;   // H*C

typedef __attribute__((ext_vector_type(16))) __bf16 bf16x16;
typedef __attribute__((ext_vector_type(2)))  __bf16 bf16x2;
typedef __attribute__((ext_vector_type(8)))  float  f32x8;

struct U4x2 { uint4 lo, hi; };      // 32B carrier for a bf16x16 fragment
struct U8   { unsigned v[8]; };     // 32B carrier (scalar build)

// ---------- small helpers ----------
__device__ inline float lrelu(float v) { return v > 0.f ? v : 0.2f * v; }

// monotonic float<->uint encoding so unsigned atomicMax == float max
__device__ inline unsigned fenc(float f) {
  unsigned u = __float_as_uint(f);
  return (u & 0x80000000u) ? ~u : (u | 0x80000000u);
}
__device__ inline float fdec(unsigned u) {
  return (u & 0x80000000u) ? __uint_as_float(u & 0x7FFFFFFFu) : __uint_as_float(~u);
}

__device__ inline void atomAddF(float* p, float v) {
  __hip_atomic_fetch_add(p, v, __ATOMIC_RELAXED, __HIP_MEMORY_SCOPE_AGENT);
}

__device__ inline void mac4(float4& a, float v, float4 w) {
  a.x = fmaf(v, w.x, a.x); a.y = fmaf(v, w.y, a.y);
  a.z = fmaf(v, w.z, a.z); a.w = fmaf(v, w.w, a.w);
}

__device__ inline unsigned pack_bf16(float a, float b) {
  bf16x2 p; p.x = (__bf16)a; p.y = (__bf16)b;      // -> v_cvt_pk_bf16_f32
  return __builtin_bit_cast(unsigned, p);
}

// B fragment from fragment-ordered LDS: two b128 loads, lane-stride 16B.
__device__ inline bf16x16 load_b_frag_sw(const unsigned* sWsz, int tile, int lane) {
  U4x2 t;
  t.lo = *(const uint4*)(sWsz + tile * 256 + lane * 4);        // pairs rp=0..3
  t.hi = *(const uint4*)(sWsz + tile * 256 + 128 + lane * 4);  // pairs rp=4..7
  return __builtin_bit_cast(bf16x16, t);
}

// ---------- K0a: fold attention vectors ----------
__global__ __launch_bounds__(256) void k_fold(
    const float* __restrict__ W, const float* __restrict__ W_edge,
    const float* __restrict__ att_src, const float* __restrict__ att_dst,
    const float* __restrict__ att_edge,
    float* __restrict__ wa_src, float* __restrict__ wa_dst, float* __restrict__ we_att) {
  int tid = threadIdx.x;            // 256 threads: (f, h)
  int f = tid >> 2, hh = tid & 3;
  float s1 = 0.f, s2 = 0.f, s3 = 0.f;
#pragma unroll 8
  for (int c = 0; c < 64; ++c) {
    float w  = W[f * HC + hh * 64 + c];
    float we = W_edge[f * HC + hh * 64 + c];
    s1 = fmaf(w,  att_src[hh * 64 + c], s1);
    s2 = fmaf(w,  att_dst[hh * 64 + c], s2);
    s3 = fmaf(we, att_edge[hh * 64 + c], s3);
  }
  wa_src[f * 4 + hh] = s1;
  wa_dst[f * 4 + hh] = s2;
  we_att[f * 4 + hh] = s3;
}

// ---------- K0b: pre-swizzle W (64x256) and lin_W (256x64) into fragment order ----------
// Tile = 32x16 (KxN). Output u32 index: tile*256 + (rp>>2)*128 + lane*4 + (rp&3),
// where lane = half*16 + n, and the u32 packs B[k0+half*8+(rp>>2)*16+2(rp&3)][n], B[..+1][n].
__global__ __launch_bounds__(256) void k_preswizzle(
    const float* __restrict__ W, const float* __restrict__ linW,
    unsigned* __restrict__ wszW, unsigned* __restrict__ wszL) {
  int t = blockIdx.x * 256 + threadIdx.x;   // 0..16383
  int m = t >> 13;                          // 0 -> W, 1 -> linW
  int w = t & 8191;
  int rp   = w & 7;
  int lane = (w >> 3) & 31;
  int tile = w >> 8;                        // 0..31
  int half = lane >> 4, n = lane & 15;
  const float* src; unsigned* dst; int Ncols, ks, ct;
  if (m == 0) { src = W;    dst = wszW; Ncols = 256; ks = tile >> 4; ct = tile & 15; }
  else        { src = linW; dst = wszL; Ncols = 64;  ks = tile >> 2; ct = tile & 3;  }
  int k   = ks * 32 + half * 8 + (rp >> 2) * 16 + 2 * (rp & 3);
  int col = ct * 16 + n;
  float a = src[(size_t)k * Ncols + col];
  float b = src[(size_t)(k + 1) * Ncols + col];
  dst[tile * 256 + (rp >> 2) * 128 + lane * 4 + (rp & 3)] = pack_bf16(a, b);
}

// ---------- K1: x = h @ W via WMMA bf16 ----------
// 128 threads = 4 waves; block = one 16-row strip; wave w covers cols [w*64, w*64+64).
__global__ __launch_bounds__(128) void k_x_gemm(
    const float* __restrict__ h, const unsigned* __restrict__ wszW, float* __restrict__ x) {
  __shared__ __align__(16) unsigned sW[8192];    // 32 KB fragment-ordered bf16 W
  __shared__ __align__(16) __bf16   sA[16 * 72]; // bf16 h tile, row pad 72
  const int tid = threadIdx.x;
  const int row0 = blockIdx.x * 16;

  {  // stage pre-swizzled W: pure b128 copy
    const uint4* g = (const uint4*)wszW;
    uint4* s = (uint4*)sW;
    for (int i = tid; i < 2048; i += 128) s[i] = g[i];
  }
  {  // stage h tile as bf16 (contiguous source: rows row0..row0+15)
    unsigned* sAu = (unsigned*)sA;
    for (int p = tid; p < 512; p += 128) {
      float2 v = *(const float2*)&h[(size_t)row0 * 64 + p * 2];
      sAu[(p >> 5) * 36 + (p & 31)] = pack_bf16(v.x, v.y);
    }
  }
  __syncthreads();

  const int wave = tid >> 5, lane = tid & 31;
  const int half = lane >> 4, mrow = lane & 15;
  f32x8 acc[4] = {};

#pragma unroll
  for (int ks = 0; ks < 2; ++ks) {
    const int k0 = ks * 32;
    const __bf16* arow = sA + mrow * 72;
    U4x2 at;
    at.lo = *(const uint4*)(arow + k0 + half * 8);        // elems 0..7  (K run 1)
    at.hi = *(const uint4*)(arow + k0 + 16 + half * 8);   // elems 8..15 (K run 2)
    bf16x16 afrag = __builtin_bit_cast(bf16x16, at);
#pragma unroll
    for (int ct = 0; ct < 4; ++ct) {
      bf16x16 bfrag = load_b_frag_sw(sW, ks * 16 + wave * 4 + ct, lane);
      acc[ct] = __builtin_amdgcn_wmma_f32_16x16x32_bf16(
          false, afrag, false, bfrag, (short)0, acc[ct], false, false);
    }
  }

  const int n0 = wave * 64, n = lane & 15;
#pragma unroll
  for (int ct = 0; ct < 4; ++ct)
#pragma unroll
    for (int v = 0; v < 8; ++v)
      x[(size_t)(row0 + half * 8 + v) * HC + n0 + ct * 16 + n] = acc[ct][v];
}

// ---------- K1b: a_src / a_dst (tiny, VALU) ----------
__global__ __launch_bounds__(256) void k_node_att(
    const float* __restrict__ h, const float* __restrict__ wa_src,
    const float* __restrict__ wa_dst, float* __restrict__ a_src, float* __restrict__ a_dst) {
  __shared__ float4 sS[64], sD[64];
  int tid = threadIdx.x;
  if (tid < 64) { sS[tid] = ((const float4*)wa_src)[tid]; sD[tid] = ((const float4*)wa_dst)[tid]; }
  __syncthreads();
  int n = blockIdx.x * 256 + tid;
  if (n >= N_NODES) return;
  const float4* row = (const float4*)(h + (size_t)n * 64);
  float4 as = {0.f, 0.f, 0.f, 0.f}, ad = {0.f, 0.f, 0.f, 0.f};
#pragma unroll
  for (int i = 0; i < 16; ++i) {
    float4 v = row[i];
    mac4(as, v.x, sS[4 * i + 0]); mac4(ad, v.x, sD[4 * i + 0]);
    mac4(as, v.y, sS[4 * i + 1]); mac4(ad, v.y, sD[4 * i + 1]);
    mac4(as, v.z, sS[4 * i + 2]); mac4(ad, v.z, sD[4 * i + 2]);
    mac4(as, v.w, sS[4 * i + 3]); mac4(ad, v.w, sD[4 * i + 3]);
  }
  ((float4*)a_src)[n] = as;
  ((float4*)a_dst)[n] = ad;
}

// ---------- K2: edge alpha (folded edge-linear) + leaky relu + segment max ----------
__global__ __launch_bounds__(256) void k_edge_alpha(
    const float* __restrict__ edge_attr, const int* __restrict__ ei,
    const float* __restrict__ a_src, const float* __restrict__ a_dst,
    const float* __restrict__ we_att, float* __restrict__ alpha_buf,
    unsigned* __restrict__ m_enc) {
  __shared__ float4 sWe[64];
  int tid = threadIdx.x;
  if (tid < 64) sWe[tid] = ((const float4*)we_att)[tid];
  __syncthreads();
  int e = blockIdx.x * 256 + tid;      // grid exact: E/256
  if (e + 8192 < N_EDGES)              // stream-ahead hint -> global_prefetch_b8
    __builtin_prefetch(edge_attr + (size_t)(e + 8192) * 64, 0, 1);
  const float4* row = (const float4*)(edge_attr + (size_t)e * 64);
  float4 acc = {0.f, 0.f, 0.f, 0.f};
#pragma unroll
  for (int i = 0; i < 16; ++i) {
    float4 v = row[i];
    mac4(acc, v.x, sWe[4 * i + 0]);
    mac4(acc, v.y, sWe[4 * i + 1]);
    mac4(acc, v.z, sWe[4 * i + 2]);
    mac4(acc, v.w, sWe[4 * i + 3]);
  }
  int s = ei[e], d = ei[N_EDGES + e];
  float4 as = ((const float4*)a_src)[s];
  float4 ad = ((const float4*)a_dst)[d];
  float4 al;
  al.x = lrelu(as.x + ad.x + acc.x);
  al.y = lrelu(as.y + ad.y + acc.y);
  al.z = lrelu(as.z + ad.z + acc.z);
  al.w = lrelu(as.w + ad.w + acc.w);
  ((float4*)alpha_buf)[e] = al;
  unsigned* mb = m_enc + (size_t)d * 4;
  atomicMax(mb + 0, fenc(al.x));
  atomicMax(mb + 1, fenc(al.y));
  atomicMax(mb + 2, fenc(al.z));
  atomicMax(mb + 3, fenc(al.w));
}

// ---------- K3: exp(alpha - m[dst]) + denom ----------
__global__ __launch_bounds__(256) void k_edge_exp(
    const int* __restrict__ ei, const unsigned* __restrict__ m_enc,
    float* __restrict__ alpha_buf, float* __restrict__ denom) {
  int e = blockIdx.x * 256 + threadIdx.x;
  int d = ei[N_EDGES + e];
  uint4 me = ((const uint4*)m_enc)[d];
  float4 raw = ((const float4*)alpha_buf)[e];
  float4 p;
  p.x = __expf(raw.x - fdec(me.x));
  p.y = __expf(raw.y - fdec(me.y));
  p.z = __expf(raw.z - fdec(me.z));
  p.w = __expf(raw.w - fdec(me.w));
  ((float4*)alpha_buf)[e] = p;
  float* db = denom + (size_t)d * 4;
  atomAddF(db + 0, p.x);
  atomAddF(db + 1, p.y);
  atomAddF(db + 2, p.z);
  atomAddF(db + 3, p.w);
}

// ---------- K4: message scatter; one wave32 per edge ----------
__global__ __launch_bounds__(256) void k_scatter(
    const float* __restrict__ x, const float* __restrict__ alpha_buf,
    const float* __restrict__ denom, const int* __restrict__ ei,
    float* __restrict__ agg) {
  int gw = (blockIdx.x * 256 + threadIdx.x) >> 5;  // edge id (grid exact)
  int lane = threadIdx.x & 31;
  int s = ei[gw], d = ei[N_EDGES + gw];
#pragma unroll
  for (int hh = 0; hh < H_HEADS; ++hh) {
    float w = alpha_buf[(size_t)gw * 4 + hh] / (denom[(size_t)d * 4 + hh] + 1e-16f);
    float2 v = *(const float2*)&x[(size_t)s * HC + hh * 64 + lane * 2];
    float* o = &agg[(size_t)d * HC + hh * 64 + lane * 2];
    atomAddF(o + 0, w * v.x);
    atomAddF(o + 1, w * v.y);
  }
}

// ---------- K5: out = (agg + bias) @ lin_W + lin_b, then LayerNorm (fused) ----------
// 256 threads = 8 waves; each wave does one 16-row strip (all 64 out cols).
__global__ __launch_bounds__(256) void k_out_gemm_ln(
    const float* __restrict__ agg, const float* __restrict__ gat_bias,
    const unsigned* __restrict__ wszL, const float* __restrict__ lin_b,
    const float* __restrict__ ln_g, const float* __restrict__ ln_beta,
    float* __restrict__ out) {
  __shared__ __align__(16) unsigned sB[8192];  // 32 KB; reused as f32 slab after barrier
  __shared__ __align__(16) float sBias[256];
  __shared__ float sLinB[64], sG[64], sBeta[64];
  const int tid = threadIdx.x;

  {  // stage pre-swizzled lin_W: pure b128 copy
    const uint4* g = (const uint4*)wszL;
    uint4* s = (uint4*)sB;
    for (int i = tid; i < 2048; i += 256) s[i] = g[i];
  }
  sBias[tid] = gat_bias[tid];
  if (tid < 64) { sLinB[tid] = lin_b[tid]; sG[tid] = ln_g[tid]; sBeta[tid] = ln_beta[tid]; }
  __syncthreads();

  const int wave = tid >> 5, lane = tid & 31;
  const int half = lane >> 4, mrow = lane & 15;
  const int strip = blockIdx.x * 8 + wave;
  const bool active = strip < (N_NODES / 16);
  const int row0 = strip * 16;
  f32x8 acc[4] = {};

  if (active) {
    const float* rowp = agg + (size_t)(row0 + mrow) * HC;
#pragma unroll
    for (int ks = 0; ks < 8; ++ks) {
      const int k0 = ks * 32;
      const int ka = k0 + half * 8;          // first K run
      float4 p0 = *(const float4*)(rowp + ka);
      float4 p1 = *(const float4*)(rowp + ka + 4);
      float4 q0 = *(const float4*)(rowp + ka + 16);
      float4 q1 = *(const float4*)(rowp + ka + 20);
      float4 b0 = *(const float4*)(sBias + ka);
      float4 b1 = *(const float4*)(sBias + ka + 4);
      float4 c0 = *(const float4*)(sBias + ka + 16);
      float4 c1 = *(const float4*)(sBias + ka + 20);
      U8 t;
      t.v[0] = pack_bf16(p0.x + b0.x, p0.y + b0.y);
      t.v[1] = pack_bf16(p0.z + b0.z, p0.w + b0.w);
      t.v[2] = pack_bf16(p1.x + b1.x, p1.y + b1.y);
      t.v[3] = pack_bf16(p1.z + b1.z, p1.w + b1.w);
      t.v[4] = pack_bf16(q0.x + c0.x, q0.y + c0.y);
      t.v[5] = pack_bf16(q0.z + c0.z, q0.w + c0.w);
      t.v[6] = pack_bf16(q1.x + c1.x, q1.y + c1.y);
      t.v[7] = pack_bf16(q1.z + c1.z, q1.w + c1.w);
      bf16x16 afrag = __builtin_bit_cast(bf16x16, t);
#pragma unroll
      for (int ct = 0; ct < 4; ++ct) {
        bf16x16 bfrag = load_b_frag_sw(sB, ks * 4 + ct, lane);
        acc[ct] = __builtin_amdgcn_wmma_f32_16x16x32_bf16(
            false, afrag, false, bfrag, (short)0, acc[ct], false, false);
      }
    }
  }
  __syncthreads();  // all waves done reading sB; reuse as per-wave f32 slab

  float* slab = (float*)sB;
  float* my = slab + wave * (16 * 64);
  if (active) {
    const int n = lane & 15;
#pragma unroll
    for (int ct = 0; ct < 4; ++ct)
#pragma unroll
      for (int v = 0; v < 8; ++v)
        my[(half * 8 + v) * 64 + ct * 16 + n] = acc[ct][v] + sLinB[ct * 16 + n];

    // wave-local LayerNorm: 2 lanes per row (32 elems each), shfl_xor combine
    const int row = lane >> 1, off = (lane & 1) * 32;
    float s = 0.f, s2 = 0.f;
#pragma unroll 8
    for (int j = 0; j < 32; ++j) {
      float v = my[row * 64 + off + j];
      s += v; s2 = fmaf(v, v, s2);
    }
    s  += __shfl_xor(s, 1, 32);
    s2 += __shfl_xor(s2, 1, 32);
    float mu  = s * (1.f / 64.f);
    float var = s2 * (1.f / 64.f) - mu * mu;
    float rstd = rsqrtf(var + 1e-5f);
#pragma unroll 8
    for (int j = 0; j < 32; ++j) {
      int col = off + j;
      float v = my[row * 64 + col];
      out[(size_t)(row0 + row) * 64 + col] = (v - mu) * rstd * sG[col] + sBeta[col];
    }
  }
}

// ---------- launcher ----------
extern "C" void kernel_launch(void* const* d_in, const int* in_sizes, int n_in,
                              void* d_out, int out_size, void* d_ws, size_t ws_size,
                              hipStream_t stream) {
  (void)in_sizes; (void)n_in; (void)out_size; (void)ws_size;
  const float* h         = (const float*)d_in[0];
  const float* edge_attr = (const float*)d_in[1];
  const int*   ei        = (const int*)d_in[2];
  const float* W         = (const float*)d_in[3];
  const float* W_edge    = (const float*)d_in[4];
  const float* att_src   = (const float*)d_in[5];
  const float* att_dst   = (const float*)d_in[6];
  const float* att_edge  = (const float*)d_in[7];
  const float* gat_bias  = (const float*)d_in[8];
  const float* lin_W     = (const float*)d_in[9];
  const float* lin_b     = (const float*)d_in[10];
  const float* ln_g      = (const float*)d_in[11];
  const float* ln_b      = (const float*)d_in[12];
  float* out = (float*)d_out;

  char* p = (char*)d_ws;
  float*    x      = (float*)p;    p += (size_t)N_NODES * HC * 4;
  float*    agg    = (float*)p;    p += (size_t)N_NODES * HC * 4;
  float*    alpha  = (float*)p;    p += (size_t)N_EDGES * H_HEADS * 4;
  float*    a_src  = (float*)p;    p += (size_t)N_NODES * H_HEADS * 4;
  float*    a_dst  = (float*)p;    p += (size_t)N_NODES * H_HEADS * 4;
  unsigned* m_enc  = (unsigned*)p; p += (size_t)N_NODES * H_HEADS * 4;
  float*    denom  = (float*)p;    p += (size_t)N_NODES * H_HEADS * 4;
  float*    wa_src = (float*)p;    p += F_DIM * H_HEADS * 4;
  float*    wa_dst = (float*)p;    p += F_DIM * H_HEADS * 4;
  float*    we_att = (float*)p;    p += F_DIM * H_HEADS * 4;
  unsigned* wszW   = (unsigned*)p; p += 8192 * 4;   // fragment-ordered bf16 W
  unsigned* wszL   = (unsigned*)p; p += 8192 * 4;   // fragment-ordered bf16 lin_W

  hipMemsetAsync(m_enc, 0, (size_t)N_NODES * H_HEADS * 4, stream);  // enc lower bound
  hipMemsetAsync(denom, 0, (size_t)N_NODES * H_HEADS * 4, stream);
  hipMemsetAsync(agg,   0, (size_t)N_NODES * HC * 4, stream);

  k_fold<<<1, 256, 0, stream>>>(W, W_edge, att_src, att_dst, att_edge, wa_src, wa_dst, we_att);
  k_preswizzle<<<64, 256, 0, stream>>>(W, lin_W, wszW, wszL);
  k_x_gemm<<<N_NODES / 16, 128, 0, stream>>>(h, wszW, x);            // 3125 blocks
  k_node_att<<<(N_NODES + 255) / 256, 256, 0, stream>>>(h, wa_src, wa_dst, a_src, a_dst);
  k_edge_alpha<<<N_EDGES / 256, 256, 0, stream>>>(edge_attr, ei, a_src, a_dst, we_att, alpha, m_enc);
  k_edge_exp<<<N_EDGES / 256, 256, 0, stream>>>(ei, m_enc, alpha, denom);
  k_scatter<<<N_EDGES / 8, 256, 0, stream>>>(x, alpha, denom, ei, agg);  // 1 wave/edge
  k_out_gemm_ln<<<(N_NODES / 16 + 7) / 8, 256, 0, stream>>>(agg, gat_bias, wszL, lin_b, ln_g, ln_b, out);
}
